// Lstm_model_26809185861928
// MI455X (gfx1250) — matrix operand order repaired and checked by reading, not compiled, not run on hardware
//
#include <hip/hip_runtime.h>
#include <stdint.h>

#define T 32768
#define EMBD 200
#define KPAD0 224
#define HID 32
#define G4 128   // 4*H
#define NTAG 5
#define TSTART 5
#define TEND 6

typedef _Float16 v16h __attribute__((ext_vector_type(16)));
typedef _Float16 v8h  __attribute__((ext_vector_type(8)));
typedef float    v8f  __attribute__((ext_vector_type(8)));

// ---------------- embedding gather + f32->f16 cast (K padded to 224) -------
__global__ void gather_cast_kernel(const int* __restrict__ sent,
                                   const float* __restrict__ emb,
                                   _Float16* __restrict__ x0) {
  long i = (long)blockIdx.x * blockDim.x + threadIdx.x;
  if (i >= (long)T * KPAD0) return;
  int t = (int)(i / KPAD0);
  int k = (int)(i % KPAD0);
  if (k == 0) {
    int tn = (t + 16 < T) ? (t + 16) : t;
    __builtin_prefetch(&emb[(long)sent[tn] * EMBD], 0, 0);  // global_prefetch_b8
  }
  _Float16 v = (_Float16)0.f;
  if (k < EMBD) v = (_Float16)emb[(long)sent[t] * EMBD + k];
  x0[i] = v;
}

// ---------------- weight f32->f16 convert with K padding -------------------
__global__ void cvt_pad_kernel(const float* __restrict__ src,
                               _Float16* __restrict__ dst,
                               int rows, int ksrc, int kdst) {
  long i = (long)blockIdx.x * blockDim.x + threadIdx.x;
  if (i >= (long)rows * kdst) return;
  int r = (int)(i / kdst);
  int k = (int)(i % kdst);
  dst[i] = (k < ksrc) ? (_Float16)src[(long)r * ksrc + k] : (_Float16)0.f;
}

__global__ void bias_sum_kernel(const float* __restrict__ a,
                                const float* __restrict__ b,
                                float* __restrict__ o, int n) {
  int i = blockIdx.x * blockDim.x + threadIdx.x;
  if (i < n) o[i] = a[i] + b[i];
}

// ---------------- WMMA input projection: pre = X @ Wih^T + bias ------------
// X: [T][lda] f16 row-major, W: [2][128][ldw] f16 row-major,
// pre: [2][T][128] f32. N dimension = dir*128+g, tiled 16 wide.
// One wave computes one 16x16 tile; K loop of 16x16x32 WMMA steps.
__global__ void __launch_bounds__(256)
wmma_pre_kernel(const _Float16* __restrict__ A, int lda,
                const _Float16* __restrict__ W, int ldw,
                const float* __restrict__ bias,
                float* __restrict__ pre, int ksteps) {
  int wave = (blockIdx.x * blockDim.x + threadIdx.x) >> 5;
  int lane = threadIdx.x & 31;
  int tileM = wave >> 4;          // 2048 row tiles of 16
  int tileN = wave & 15;          // 16 col tiles of 16 (2 dirs x 128 cols)
  int hi = lane >> 4;             // lane group (K offset select)
  int lo = lane & 15;             // M index (A) / N index (B,D)
  int trow = tileM * 16 + lo;
  int ncol = tileN * 16 + lo;
  int dir = ncol >> 7;
  int g   = ncol & 127;
  const _Float16* Arow = A + (long)trow * lda;
  const _Float16* Wrow = W + ((long)dir * G4 + g) * ldw;

  v8f c = {};
  for (int ks = 0; ks < ksteps; ++ks) {
    int k0 = ks * 32;
    // A fragment (ISA 16-bit A 16x32 layout): two contiguous 8-half chunks
    v8h alo = *(const v8h*)(Arow + k0 + 8 * hi);
    v8h ahi = *(const v8h*)(Arow + k0 + 16 + 8 * hi);
    v16h a = __builtin_shufflevector(alo, ahi,
                                     0, 1, 2, 3, 4, 5, 6, 7,
                                     8, 9, 10, 11, 12, 13, 14, 15);
    // B fragment: same K order as A (two 8-half chunks)
    v8h blo = *(const v8h*)(Wrow + k0 + 8 * hi);
    v8h bhi = *(const v8h*)(Wrow + k0 + 16 + 8 * hi);
    v16h b = __builtin_shufflevector(blo, bhi,
                                     0, 1, 2, 3, 4, 5, 6, 7,
                                     8, 9, 10, 11, 12, 13, 14, 15);
    c = __builtin_amdgcn_wmma_f32_16x16x32_f16(
        /*neg_a=*/false, a, /*neg_b=*/false, b,
        /*c_mod=*/(short)0, c, /*reuse_a=*/false, /*reuse_b=*/false);
  }
  float bv = bias[dir * G4 + g];
  long base = (long)dir * T * G4;
  #pragma unroll
  for (int r = 0; r < 8; ++r) {
    int m = r + 8 * hi;           // ISA 32-bit C/D 16x16 layout
    pre[base + (long)(tileM * 16 + m) * G4 + g] = c[r] + bv;
  }
}

// ---------------- sequential LSTM scan, one block per direction ------------
// pre: [2][T][128], Whh: [2][128][32]. h in LDS, Whh row in registers.
__global__ void __launch_bounds__(128)
lstm_scan_kernel(const float* __restrict__ pre,
                 const float* __restrict__ Whh,
                 _Float16* __restrict__ out16,   // [T][64] (nullable)
                 float* __restrict__ out32) {    // [T][64] (nullable)
  int dir = blockIdx.x;
  int tid = threadIdx.x;          // gate row 0..127
  __shared__ float s_h[HID];
  __shared__ float s_act[G4];
  float w[HID];
  #pragma unroll
  for (int k = 0; k < HID; ++k)
    w[k] = Whh[((long)dir * G4 + tid) * HID + k];
  float c = 0.f;
  if (tid < HID) s_h[tid] = 0.f;
  __syncthreads();
  const float* preD = pre + (long)dir * T * G4;
  for (int step = 0; step < T; ++step) {
    int t = dir ? (T - 1 - step) : step;
    float gacc = preD[(long)t * G4 + tid];
    #pragma unroll
    for (int k = 0; k < HID; ++k) gacc += w[k] * s_h[k];
    float a;
    if (tid >= 64 && tid < 96) a = tanhf(gacc);          // g gate
    else a = 1.f / (1.f + __expf(-gacc));                // i,f,o gates
    s_act[tid] = a;
    __syncthreads();
    if (tid < HID) {
      float ig = s_act[tid], fg = s_act[HID + tid];
      float gg = s_act[2 * HID + tid], og = s_act[3 * HID + tid];
      c = fg * c + ig * gg;
      float h = og * tanhf(c);
      s_h[tid] = h;
      long o = (long)t * (2 * HID) + dir * HID + tid;
      if (out16) out16[o] = (_Float16)h;
      if (out32) out32[o] = h;
    }
    __syncthreads();
  }
}

// ---------------- logits: l = x2 @ W_lin^T + b_lin (64 -> 5) ---------------
__global__ void logits_kernel(const float* __restrict__ x,
                              const float* __restrict__ Wl,
                              const float* __restrict__ bl,
                              float* __restrict__ l) {
  int t = blockIdx.x * blockDim.x + threadIdx.x;
  if (t >= T) return;
  const float* xr = x + (long)t * 64;
  #pragma unroll
  for (int j = 0; j < NTAG; ++j) {
    float acc = bl[j];
    #pragma unroll 8
    for (int k = 0; k < 64; ++k) acc += xr[k] * Wl[j * 64 + k];
    l[(long)t * NTAG + j] = acc;
  }
}

// ---------------- Viterbi: forward max-scan + LDS-resident backtrace -------
// All (T-1)*5 backpointers live in 160 KB of dynamic LDS (CDNA5: 320 KB/WGP).
__global__ void viterbi_kernel(const float* __restrict__ l,
                               const float* __restrict__ trans,
                               float* __restrict__ out) {
  extern __shared__ unsigned char s_bp[];   // (T-1)*NTAG bytes
  __shared__ float s_prev[NTAG];
  __shared__ float s_next[NTAG];
  int tid = threadIdx.x;
  float trn[NTAG];
  if (tid < NTAG) {
    #pragma unroll
    for (int i = 0; i < NTAG; ++i) trn[i] = trans[i * 7 + tid];
    s_prev[tid] = l[tid] + trans[TSTART * 7 + tid];
  }
  __syncthreads();
  for (int t = 1; t < T; ++t) {
    if (tid < NTAG) {
      float best = s_prev[0] + trn[0]; int bi = 0;
      #pragma unroll
      for (int i = 1; i < NTAG; ++i) {
        float v = s_prev[i] + trn[i];
        if (v > best) { best = v; bi = i; }
      }
      s_bp[(t - 1) * NTAG + tid] = (unsigned char)bi;
      s_next[tid] = best + l[(long)t * NTAG + tid];
    }
    __syncthreads();
    if (tid < NTAG) s_prev[tid] = s_next[tid];
    __syncthreads();
  }
  if (tid == 0) {
    float best = s_prev[0] + trans[0 * 7 + TEND]; int bi = 0;
    for (int j = 1; j < NTAG; ++j) {
      float v = s_prev[j] + trans[j * 7 + TEND];
      if (v > best) { best = v; bi = j; }
    }
    out[0] = best;
    int idx = bi;
    out[1 + (T - 1)] = (float)idx;
    for (int t = T - 2; t >= 0; --t) {
      idx = s_bp[t * NTAG + idx];
      out[1 + t] = (float)idx;
    }
  }
}

extern "C" void kernel_launch(void* const* d_in, const int* in_sizes, int n_in,
                              void* d_out, int out_size, void* d_ws, size_t ws_size,
                              hipStream_t stream) {
  const int*   sent  = (const int*)d_in[0];
  const float* emb   = (const float*)d_in[1];
  const float* Wih0  = (const float*)d_in[2];
  const float* Whh0  = (const float*)d_in[3];
  const float* bih0  = (const float*)d_in[4];
  const float* bhh0  = (const float*)d_in[5];
  const float* Wih1  = (const float*)d_in[6];
  const float* Whh1  = (const float*)d_in[7];
  const float* bih1  = (const float*)d_in[8];
  const float* bhh1  = (const float*)d_in[9];
  const float* Wlin  = (const float*)d_in[10];
  const float* blin  = (const float*)d_in[11];
  const float* trans = (const float*)d_in[12];
  float* out = (float*)d_out;
  (void)in_sizes; (void)n_in; (void)out_size; (void)ws_size;

  char* ws = (char*)d_ws;
  size_t off = 0;
  auto alloc = [&](size_t bytes) -> void* {
    void* p = ws + off;
    off = (off + bytes + 255) & ~(size_t)255;
    return p;
  };
  _Float16* x0  = (_Float16*)alloc((size_t)T * KPAD0 * 2);      // 14.7 MB
  _Float16* w0f = (_Float16*)alloc((size_t)2 * G4 * KPAD0 * 2); // 112 KB
  _Float16* w1f = (_Float16*)alloc((size_t)2 * G4 * 64 * 2);    // 32 KB
  float*    b0  = (float*)alloc((size_t)2 * G4 * 4);
  float*    b1  = (float*)alloc((size_t)2 * G4 * 4);
  float*    pre = (float*)alloc((size_t)2 * T * G4 * 4);        // 33.5 MB (reused)
  _Float16* x1  = (_Float16*)alloc((size_t)T * 64 * 2);         // 4.2 MB
  float*    x2  = (float*)alloc((size_t)T * 64 * 4);            // 8.4 MB
  float*    lg  = (float*)alloc((size_t)T * NTAG * 4);          // 0.66 MB

  // weight prep
  hipLaunchKernelGGL(cvt_pad_kernel, dim3((2 * G4 * KPAD0 + 255) / 256), dim3(256),
                     0, stream, Wih0, w0f, 2 * G4, EMBD, KPAD0);
  hipLaunchKernelGGL(cvt_pad_kernel, dim3((2 * G4 * 64 + 255) / 256), dim3(256),
                     0, stream, Wih1, w1f, 2 * G4, 64, 64);
  hipLaunchKernelGGL(bias_sum_kernel, dim3(1), dim3(256), 0, stream, bih0, bhh0, b0, 2 * G4);
  hipLaunchKernelGGL(bias_sum_kernel, dim3(1), dim3(256), 0, stream, bih1, bhh1, b1, 2 * G4);

  // embedding gather + cast
  long ng = (long)T * KPAD0;
  hipLaunchKernelGGL(gather_cast_kernel, dim3((unsigned)((ng + 255) / 256)), dim3(256),
                     0, stream, sent, emb, x0);

  // layer 0: WMMA input projection, then sequential scan (fwd/bwd concurrent)
  hipLaunchKernelGGL(wmma_pre_kernel, dim3(4096), dim3(256), 0, stream,
                     x0, KPAD0, w0f, KPAD0, b0, pre, KPAD0 / 32);
  hipLaunchKernelGGL(lstm_scan_kernel, dim3(2), dim3(128), 0, stream,
                     pre, Whh0, x1, (float*)nullptr);

  // layer 1: WMMA input projection (K=64), then scan
  hipLaunchKernelGGL(wmma_pre_kernel, dim3(4096), dim3(256), 0, stream,
                     x1, 64, w1f, 64, b1, pre, 2);
  hipLaunchKernelGGL(lstm_scan_kernel, dim3(2), dim3(128), 0, stream,
                     pre, Whh1, (_Float16*)nullptr, x2);

  // logits + Viterbi (backpointers in 160 KB dynamic LDS)
  hipLaunchKernelGGL(logits_kernel, dim3((T + 255) / 256), dim3(256), 0, stream,
                     x2, Wlin, blin, lg);
  hipLaunchKernelGGL(viterbi_kernel, dim3(1), dim3(32),
                     (size_t)T * NTAG, stream, lg, trans, out);
}
